// SimpleARRQSFlow_20572893348387
// MI455X (gfx1250) — matrix operand order repaired and checked by reading, not compiled, not run on hardware
//
#include <hip/hip_runtime.h>
#include <hip/hip_bf16.h>
#include <stdint.h>

#define DEV __device__ __forceinline__

typedef __attribute__((ext_vector_type(16))) __bf16 v16bf;
typedef __attribute__((ext_vector_type(8)))  float  v8f;
typedef int v4i __attribute__((vector_size(16)));   // matches async-LDS builtin param

#define AS1 __attribute__((address_space(1)))
#define AS3 __attribute__((address_space(3)))

#if __has_builtin(__builtin_amdgcn_global_load_async_to_lds_b128) && \
    __has_builtin(__builtin_amdgcn_s_wait_asynccnt)
#define USE_ASYNC_LDS 1
#else
#define USE_ASYNC_LDS 0
#endif

union FragAB { uint4 q[2]; v16bf v; };

static constexpr int BATCH = 4096;
static constexpr int S1   = 15;     // NB_SITES - 1
static constexpr int HID  = 512;
static constexpr int DPAD = 128;    // D=120 padded to 128 for K-alignment
static constexpr int OUTC = 99;     // 3*K + 3
static constexpr int OUTP = 100;    // padded param row stride

DEV unsigned short f2bf(float f) {
  unsigned u = __float_as_uint(f);
  unsigned r = ((u >> 16) & 1u) + 0x7fffu;     // round-to-nearest-even
  return (unsigned short)((u + r) >> 16);
}
DEV float bf2f(unsigned short h) { return __uint_as_float(((unsigned)h) << 16); }

DEV float gelu_tanh(float v) {
  float u = 0.7978845608028654f * (v + 0.044715f * v * v * v);
  return 0.5f * v * (1.0f + tanhf(u));
}

DEV float invsymlu(float v) { return (v >= 0.f) ? (v + 1.f) : (1.f / (1.f - v)); }

// ---------------------------------------------------------------------------
// Weight prep: fp32 [s][k][n] -> bf16 transposed [s][Np][Kp], optional AR mask
// ---------------------------------------------------------------------------
__global__ void prep_w(const float* __restrict__ W, unsigned short* __restrict__ Wt,
                       int S, int Kd, int N, int Kp, int Np, int maskMul) {
  long total = (long)S * Np * Kp;
  for (long i = (long)blockIdx.x * 256 + threadIdx.x; i < total;
       i += (long)gridDim.x * 256) {
    int k = (int)(i % Kp);
    long r = i / Kp;
    int n = (int)(r % Np);
    int s = (int)(r / Np);
    float v = 0.f;
    if (k < Kd && n < N && (maskMul <= 0 || k < maskMul * (s + 1)))
      v = W[((long)s * Kd + k) * N + n];
    Wt[i] = f2bf(v);
  }
}

// ---------------------------------------------------------------------------
// Frequency embedding: x[:, :15] -> e bf16 [B][128] (cols 120..127 zero)
// ---------------------------------------------------------------------------
__global__ void embed_kernel(const float* __restrict__ x, unsigned short* __restrict__ e) {
  int i = blockIdx.x * 256 + threadIdx.x;
  if (i >= BATCH * DPAD) return;
  int b = i >> 7, c = i & 127;
  float v = 0.f;
  if (c < 120) {
    int s0 = c >> 3, f = c & 7;
    float xv = x[b * 16 + s0];
    int j = f & 3;                                  // div = 100^{-j/4}
    float div = __expf(-0.25f * (float)j * logf(100.f));
    float arg = xv * div;
    v = (f < 4) ? __sinf(arg) : __cosf(arg);
  }
  e[i] = f2bf(v);
}

// ---------------------------------------------------------------------------
// Fused GEMM: C = A(bf16) @ Wt(bf16,[n][k]) via V_WMMA_F32_16X16X32_BF16
//   + bias (+residual) + gelu + layernorm  -> bf16 act     (!FINAL)
//   + bias                                 -> f32 params   ( FINAL)
// block = 256 thr (8 wave32), tile M=32, N=NDIM, K staged 32-deep in LDS.
// Stage buffers are uint4-typed so fragment access lowers to ds_load_b128;
// staging uses GLOBAL_LOAD_ASYNC_TO_LDS_B128 (ASYNCcnt) when available.
// ---------------------------------------------------------------------------
template <int KDIM, int NDIM, bool RESID, bool FINAL>
__global__ __launch_bounds__(256)
void gemm_fused(const unsigned short* __restrict__ Aact, long aSiteStride,
                const unsigned short* __restrict__ Wt,
                const float* __restrict__ bias,
                const float* __restrict__ lnS, const float* __restrict__ lnB,
                unsigned short* __restrict__ Out, float* __restrict__ ParamsOut) {
  static_assert(!RESID || KDIM == NDIM, "residual needs square layer");
  constexpr int KP4 = 5;               // 5 x uint4 per row = 40 halves (80B, pad vs banks)
  constexpr int TPW = NDIM / 64;       // 16x16 tiles per wave
  struct Stage { uint4 A[32 * KP4]; uint4 Bt[NDIM * KP4]; };
  union Sm { Stage st; float C[32 * NDIM]; };
  __shared__ Sm sh;

  const int tid  = threadIdx.x;
  const int wave = tid >> 5;
  const int lane = tid & 31;
  const int s = blockIdx.y;
  const int rowBase = blockIdx.x * 32;

  const unsigned short* Ap = Aact + (long)s * aSiteStride + (long)rowBase * KDIM;
  const unsigned short* Wp = Wt + (long)s * NDIM * KDIM;

  const int mt = wave & 1;                   // row tile 0/1
  const int ncBase = (wave >> 1) * (TPW * 16);

  v8f acc[TPW] = {};

  for (int k0 = 0; k0 < KDIM; k0 += 32) {
#if USE_ASYNC_LDS
    if (tid < 128) {                          // stage A tile (32 x 32 halves)
      const int r = tid >> 2, q = tid & 3;
      __builtin_amdgcn_global_load_async_to_lds_b128(
          (AS1 v4i*)(uintptr_t)(Ap + (long)r * KDIM + k0 + q * 8),
          (AS3 v4i*)&sh.st.A[r * KP4 + q], 0, 0);
    }
    for (int i = tid; i < NDIM * 4; i += 256) {  // stage B^T tile (NDIM x 32)
      const int n = i >> 2, q = i & 3;
      __builtin_amdgcn_global_load_async_to_lds_b128(
          (AS1 v4i*)(uintptr_t)(Wp + (long)n * KDIM + k0 + q * 8),
          (AS3 v4i*)&sh.st.Bt[n * KP4 + q], 0, 0);
    }
    __builtin_amdgcn_s_wait_asynccnt(0);
#else
    if (tid < 128) {                          // stage A tile (32 x 32 halves)
      const int r = tid >> 2, q = tid & 3;
      sh.st.A[r * KP4 + q] = *(const uint4*)(Ap + (long)r * KDIM + k0 + q * 8);
    }
    for (int i = tid; i < NDIM * 4; i += 256) {  // stage B^T tile (NDIM x 32)
      const int n = i >> 2, q = i & 3;
      sh.st.Bt[n * KP4 + q] = *(const uint4*)(Wp + (long)n * KDIM + k0 + q * 8);
      if (k0 + 32 < KDIM)
        __builtin_prefetch(Wp + (long)n * KDIM + k0 + 32 + q * 8, 0, 1);
    }
#endif
    __syncthreads();

    // A fragment: lane<16 -> row=lane, K {0..7,16..23}; lane>=16 -> K {8..15,24..31}
    FragAB a;
    const int arow = mt * 16 + (lane & 15);
    const int aq   = lane >> 4;               // 0 -> halves {0..7}, 1 -> {8..15}
    a.q[0] = sh.st.A[arow * KP4 + aq];
    a.q[1] = sh.st.A[arow * KP4 + aq + 2];

#pragma unroll
    for (int t = 0; t < TPW; ++t) {
      // B fragment: lane<16 -> col=lane, K {0..7,16..23}; lane>=16 -> K {8..15,24..31}
      FragAB b;
      const int bn = ncBase + t * 16 + (lane & 15);
      const int bq = lane >> 4;               // 0 -> K {0..7,16..23}, 1 -> K {8..15,24..31}
      b.q[0] = sh.st.Bt[bn * KP4 + bq];
      b.q[1] = sh.st.Bt[bn * KP4 + bq + 2];
      acc[t] = __builtin_amdgcn_wmma_f32_16x16x32_bf16(
          false, a.v, false, b.v, (short)0, acc[t], false, false);
    }
    __syncthreads();
  }

  if constexpr (FINAL) {
    const float* bfp = bias + s * OUTC;
#pragma unroll
    for (int t = 0; t < TPW; ++t) {
      const int col = ncBase + t * 16 + (lane & 15);
      if (col < OUTC) {
        const int rl = mt * 16 + (lane >> 4) * 8;
#pragma unroll
        for (int r = 0; r < 8; ++r)
          ParamsOut[(((long)(rowBase + rl + r)) * S1 + s) * OUTP + col] =
              acc[t][r] + bfp[col];
      }
    }
  } else {
    // scatter accumulators to fp32 LDS (D layout: lane<16 rows 0..7, lane>=16 rows 8..15)
#pragma unroll
    for (int t = 0; t < TPW; ++t) {
      const int col = ncBase + t * 16 + (lane & 15);
      const int rl  = mt * 16 + (lane >> 4) * 8;
#pragma unroll
      for (int r = 0; r < 8; ++r) sh.C[(rl + r) * NDIM + col] = acc[t][r];
    }
    __syncthreads();

    // bias (+residual) + gelu, 8 columns per task (vectorized reads)
    const float* bp = bias + s * NDIM;
    constexpr int W8 = NDIM / 8;
    for (int t8 = tid; t8 < 32 * W8; t8 += 256) {
      const int r  = t8 / W8;
      const int c8 = (t8 - r * W8) * 8;
      uint4 rsd = {};
      if constexpr (RESID) rsd = *(const uint4*)(Ap + (long)r * KDIM + c8);
      const float4 bv0 = *(const float4*)(bp + c8);
      const float4 bv1 = *(const float4*)(bp + c8 + 4);
      float bb[8] = {bv0.x, bv0.y, bv0.z, bv0.w, bv1.x, bv1.y, bv1.z, bv1.w};
#pragma unroll
      for (int j = 0; j < 8; ++j) {
        float v = sh.C[r * NDIM + c8 + j] + bb[j];
        if constexpr (RESID) v += bf2f(((const unsigned short*)&rsd)[j]);
        sh.C[r * NDIM + c8 + j] = gelu_tanh(v);
      }
    }
    __syncthreads();

    // LayerNorm: one wave per row, wave32 shuffle reduction; packed bf16x2 stores
    const float* sp  = lnS + s * NDIM;
    const float* bn2 = lnB + s * NDIM;
    unsigned short* outp = Out + (long)s * ((long)BATCH * HID) + (long)rowBase * NDIM;
    for (int rr = wave; rr < 32; rr += 8) {
      float sum = 0.f, ssq = 0.f;
#pragma unroll
      for (int j = 0; j < NDIM / 32; ++j) {
        float v = sh.C[rr * NDIM + lane + j * 32];
        sum += v; ssq += v * v;
      }
      for (int off = 16; off > 0; off >>= 1) {
        sum += __shfl_xor(sum, off, 32);
        ssq += __shfl_xor(ssq, off, 32);
      }
      const float inv = 1.0f / (float)NDIM;
      float mean = sum * inv;
      float var  = ssq * inv - mean * mean;
      float rstd = rsqrtf(var + 1e-6f);
#pragma unroll
      for (int j = 0; j < NDIM / 64; ++j) {
        const int c = 2 * lane + 64 * j;
        float v0 = (sh.C[rr * NDIM + c]     - mean) * rstd * sp[c]     + bn2[c];
        float v1 = (sh.C[rr * NDIM + c + 1] - mean) * rstd * sp[c + 1] + bn2[c + 1];
        unsigned pk = (unsigned)f2bf(v0) | ((unsigned)f2bf(v1) << 16);
        *(unsigned*)(outp + (long)rr * NDIM + c) = pk;
      }
    }
  }
}

// ---------------------------------------------------------------------------
// Rational-quadratic spline, one thread per (b, site); streamed cumsums.
// ---------------------------------------------------------------------------
__global__ void rqs_kernel(const float* __restrict__ x, const float* __restrict__ bias0,
                           const float* __restrict__ params,
                           float* __restrict__ outY, float* __restrict__ ladps) {
  int i = blockIdx.x * 256 + threadIdx.x;
  if (i >= BATCH * 16) return;
  const int b = i >> 4, site = i & 15;
  const float* pr = (site == 0) ? bias0 : (params + ((long)b * S1 + (site - 1)) * OUTP);
  const float inp = x[i];

  float Wtot = 0.f, Htot = 0.f;
#pragma unroll 4
  for (int k = 0; k < 32; ++k) {
    Wtot += 0.001f + invsymlu(pr[k]);
    Htot += 0.001f + invsymlu(pr[32 + k]);
  }
  const float d0 = 0.001f + invsymlu(pr[64]);
  const float dN = 0.001f + invsymlu(pr[96]);
  const float cx = pr[97], cy = pr[98];
  const float cw0 = cx - 0.5f * Wtot, ch0 = cy - 0.5f * Htot;
  const float cwL = cw0 + Wtot,       chL = ch0 + Htot;

  float o, lad;
  if (inp < cw0) {
    o = ch0 - (cw0 - inp) * d0; lad = logf(d0);
  } else if (inp >= cwL) {
    o = (inp - cwL) * dN + chL; lad = logf(dN);
  } else {
    float cw = cw0, ch = ch0;
    float icw = cw0, ich = ch0, ibw = 1.f, ih = 1.f, ider = d0, iderp = d0;
    bool found = false;
    for (int k = 0; k < 32; ++k) {
      const float wk = 0.001f + invsymlu(pr[k]);
      const float hk = 0.001f + invsymlu(pr[32 + k]);
      const float nw = cw + wk;
      if (!found && (inp < nw || k == 31)) {
        found = true;
        icw = cw; ich = ch; ibw = wk; ih = hk;
        ider  = 0.001f + invsymlu(pr[64 + k]);
        iderp = 0.001f + invsymlu(pr[64 + k + 1]);
      }
      cw = nw; ch += hk;
    }
    const float th   = (inp - icw) / ibw;
    const float tomt = th * (1.f - th);
    const float idl  = ih / ibw;
    const float num  = ih * (idl * th * th + ider * tomt);
    const float den  = idl + (ider + iderp - 2.f * idl) * tomt;
    o = ich + num / den;
    const float omt = 1.f - th;
    const float dnum = idl * idl * (iderp * th * th + 2.f * idl * tomt + ider * omt * omt);
    lad = logf(dnum) - 2.f * logf(den);
  }
  outY[i]  = o;
  ladps[i] = lad;
}

__global__ void lad_reduce(const float* __restrict__ ladps, float* __restrict__ outLad) {
  int b = blockIdx.x * 256 + threadIdx.x;
  if (b >= BATCH) return;
  float s = 0.f;
#pragma unroll
  for (int k = 0; k < 16; ++k) s += ladps[b * 16 + k];
  outLad[b] = s;
}

// ---------------------------------------------------------------------------
extern "C" void kernel_launch(void* const* d_in, const int* in_sizes, int n_in,
                              void* d_out, int out_size, void* d_ws, size_t ws_size,
                              hipStream_t stream) {
  (void)in_sizes; (void)n_in; (void)out_size; (void)ws_size;
  const float* x     = (const float*)d_in[0];
  const float* bias0 = (const float*)d_in[1];
  const float* W0    = (const float*)d_in[2];
  const float* b0    = (const float*)d_in[3];
  const float* ln0s  = (const float*)d_in[4];
  const float* ln0b  = (const float*)d_in[5];
  const float* W1    = (const float*)d_in[6];
  const float* b1    = (const float*)d_in[7];
  const float* ln1s  = (const float*)d_in[8];
  const float* ln1b  = (const float*)d_in[9];
  const float* W2    = (const float*)d_in[10];
  const float* b2    = (const float*)d_in[11];
  const float* ln2s  = (const float*)d_in[12];
  const float* ln2b  = (const float*)d_in[13];
  const float* Wf    = (const float*)d_in[14];
  const float* bf    = (const float*)d_in[15];

  char* wsp = (char*)d_ws;
  auto alloc = [&](size_t bytes) -> char* {
    char* p = wsp; wsp += (bytes + 255) & ~(size_t)255; return p;
  };
  unsigned short* eBf  = (unsigned short*)alloc((size_t)BATCH * DPAD * 2);
  unsigned short* W0t  = (unsigned short*)alloc((size_t)S1 * HID * DPAD * 2);
  unsigned short* W1t  = (unsigned short*)alloc((size_t)S1 * HID * HID * 2);
  unsigned short* W2t  = (unsigned short*)alloc((size_t)S1 * HID * HID * 2);
  unsigned short* Wft  = (unsigned short*)alloc((size_t)S1 * DPAD * HID * 2);
  unsigned short* actA = (unsigned short*)alloc((size_t)S1 * BATCH * HID * 2);
  unsigned short* actB = (unsigned short*)alloc((size_t)S1 * BATCH * HID * 2);
  float* params        = (float*)alloc((size_t)BATCH * S1 * OUTP * 4);
  float* ladps         = (float*)alloc((size_t)BATCH * 16 * 4);

  // one-time (per call) weight prep: bf16, transposed [s][n][k], AR mask folded
  prep_w<<<3840, 256, 0, stream>>>(W0, W0t, S1, 120, HID, DPAD, HID, 8);
  prep_w<<<4096, 256, 0, stream>>>(W1, W1t, S1, HID, HID, HID, HID, 0);
  prep_w<<<4096, 256, 0, stream>>>(W2, W2t, S1, HID, HID, HID, HID, 0);
  prep_w<<<3840, 256, 0, stream>>>(Wf, Wft, S1, HID, OUTC, HID, DPAD, 0);
  embed_kernel<<<(BATCH * DPAD) / 256, 256, 0, stream>>>(x, eBf);

  dim3 g(BATCH / 32, S1);
  gemm_fused<DPAD, HID, false, false><<<g, 256, 0, stream>>>(
      eBf, 0L, W0t, b0, ln0s, ln0b, actA, nullptr);
  gemm_fused<HID, HID, true, false><<<g, 256, 0, stream>>>(
      actA, (long)BATCH * HID, W1t, b1, ln1s, ln1b, actB, nullptr);
  gemm_fused<HID, HID, true, false><<<g, 256, 0, stream>>>(
      actB, (long)BATCH * HID, W2t, b2, ln2s, ln2b, actA, nullptr);
  gemm_fused<HID, DPAD, false, true><<<g, 256, 0, stream>>>(
      actA, (long)BATCH * HID, Wft, bf, nullptr, nullptr, nullptr, params);

  float* outY   = (float*)d_out;
  float* outLad = outY + (size_t)BATCH * 16;
  rqs_kernel<<<(BATCH * 16) / 256, 256, 0, stream>>>(x, bias0, params, outY, ladps);
  lad_reduce<<<BATCH / 256, 256, 0, stream>>>(ladps, outLad);
}
